// HungarianMatcher_3126736191887
// MI455X (gfx1250) — matrix operand order repaired and checked by reading, not compiled, not run on hardware
//
#include <hip/hip_runtime.h>

typedef __attribute__((ext_vector_type(16))) _Float16 v16h;
typedef __attribute__((ext_vector_type(8)))  _Float16 v8h;
typedef __attribute__((ext_vector_type(4)))  _Float16 v4h;
typedef __attribute__((ext_vector_type(8)))  float    v8f;

#define NPRED 1200
#define MTGT  256
#define KP    16384     // 128*128 pixels
#define NCLS  80

#define LOG2E 1.44269504088896340736f
#define LN2   0.69314718055994530942f

__device__ __forceinline__ float fast_sigmoid(float x) {
  // rcp(1 + 2^(-x*log2e)) : mul + v_exp_f32 + add + v_rcp_f32
  return __builtin_amdgcn_rcpf(1.0f + __builtin_amdgcn_exp2f(-x * LOG2E));
}
__device__ __forceinline__ float neg_softplus(float x) {
  // log(1-sigmoid(x)) = -softplus(x) = -(max(x,0) + ln2*log2(1 + 2^(-|x|*log2e)))
  // log2 argument is in [1,2]: raw v_log_f32 is safe (no denorm/special cases).
  const float t = 1.0f + __builtin_amdgcn_exp2f(-fabsf(x) * LOG2E);
  return -fmaf(LN2, __builtin_amdgcn_logf(t), fmaxf(x, 0.0f));
}

// ---------------------------------------------------------------------------
// Kernel 1: target masks 256x256 -> 128x128 (half-pixel bilinear downsample
// by 2 == exact 2x2 average pool), convert to f16, and per-target pixel sum.
// ---------------------------------------------------------------------------
__global__ __launch_bounds__(256) void k_prep_tgt(const float* __restrict__ tm,
                                                  _Float16* __restrict__ T,
                                                  float* __restrict__ tsum) {
  const int m   = blockIdx.x;     // 0..255
  const int tid = threadIdx.x;    // 0..255
  const float* in = tm + (size_t)m * 65536;
  float s = 0.0f;
  for (int p = tid; p < KP; p += 256) {
    const int y = p >> 7;
    const int x = p & 127;
    const float* r0 = in + (size_t)(2 * y) * 256 + 2 * x;
    const float v = 0.25f * (r0[0] + r0[1] + r0[256] + r0[257]);
    T[(size_t)m * KP + p] = (_Float16)v;
    s += v;
  }
  __shared__ float red[256];
  red[tid] = s;
  __syncthreads();
  #pragma unroll
  for (int off = 128; off > 0; off >>= 1) {
    if (tid < off) red[tid] += red[tid + off];
    __syncthreads();
  }
  if (tid == 0) tsum[m] = red[0];
}

// ---------------------------------------------------------------------------
// Kernel 2: fused dual GEMM with WMMA.
//   C1 = sigmoid(X) @ T^T       (dice numerator)
//   C2 = X @ T^T                (BCE mask cost, since log p - log(1-p) = x)
//   ssum[n] = sum_p sigmoid(x), lsum[n] = sum_p log(1 - sigmoid(x))
// Grid: (75 m-tiles, 2 column strips). Block: 256 threads = 8 waves.
// Per outer iteration a 16x64 f32 A tile is staged to LDS (as f16 X and f16
// sigmoid(X)); each wave owns one 16-col n-tile and issues 4 WMMAs per
// barrier pair (two 32-K subtiles x two accumulators). Row sums (incl. the
// transcendental-heavy softplus) are computed only by column strip 0.
// ---------------------------------------------------------------------------
__global__ __launch_bounds__(256) void k_gemm(const float* __restrict__ pm,
                                              const _Float16* __restrict__ T,
                                              float* __restrict__ C1,
                                              float* __restrict__ C2,
                                              float* __restrict__ ssum,
                                              float* __restrict__ lsum) {
  __shared__ __align__(16) _Float16 Xh[16 * 64];
  __shared__ __align__(16) _Float16 Sh[16 * 64];
  __shared__ float redS[256];
  __shared__ float redL[256];

  const int tid    = threadIdx.x;
  const int mblk   = blockIdx.x;     // 0..74  (16 pred rows each)
  const int cstrip = blockIdx.y;     // 0..1   (128 target cols each)
  const int wave   = tid >> 5;
  const int lane   = tid & 31;

  const int n0 = (cstrip * 8 + wave) * 16;   // this wave's column base

  // A staging: thread -> (row 0..15, 4 consecutive K elements), b128 loads
  const int rowA = tid >> 4;          // 0..15
  const int colA = (tid & 15) * 4;    // 0,4,...,60
  const float* arow = pm + (size_t)(mblk * 16 + rowA) * KP + colA;

  // Fragment lane mappings (CDNA5 16-bit WMMA layouts):
  //  A: lane holds row (lane&15); elems 0..7 -> K = (lane>>4)*8 + e,
  //     elems 8..15 -> K = 16 + (lane>>4)*8 + (e-8).
  //  B: lane holds col (lane&15); same K mapping as A.
  const int fr  = lane & 15;
  const int akb = (lane >> 4) * 8;
  const int bkb = (lane >> 4) * 8;
  const _Float16* brow = T + (size_t)(n0 + fr) * KP + bkb;

  v8f acc1 = {};   // sigmoid(X) @ T^T
  v8f acc2 = {};   // X @ T^T
  float ps = 0.0f, pl = 0.0f;

  for (int k0 = 0; k0 < KP; k0 += 64) {
    // ---- stage 16x64 A tile (f32 -> f16 X and f16 sigmoid(X)) ----
    const float4 x4 = *(const float4*)(arow + k0);
    const float p0 = fast_sigmoid(x4.x);
    const float p1 = fast_sigmoid(x4.y);
    const float p2 = fast_sigmoid(x4.z);
    const float p3 = fast_sigmoid(x4.w);
    if (cstrip == 0) {   // uniform branch: row sums computed by strip 0 only
      ps += (p0 + p1) + (p2 + p3);
      pl += (neg_softplus(x4.x) + neg_softplus(x4.y)) +
            (neg_softplus(x4.z) + neg_softplus(x4.w));
    }
    const v4h xh = { (_Float16)x4.x, (_Float16)x4.y,
                     (_Float16)x4.z, (_Float16)x4.w };
    const v4h sh = { (_Float16)p0, (_Float16)p1,
                     (_Float16)p2, (_Float16)p3 };
    *(v4h*)(&Xh[rowA * 64 + colA]) = xh;
    *(v4h*)(&Sh[rowA * 64 + colA]) = sh;
    __syncthreads();

    // ---- two 32-K subtiles ----
    #pragma unroll
    for (int s = 0; s < 2; ++s) {
      const int ks = s * 32;
      v16h bf, xf, sf;
      {
        v8h* bp = (v8h*)&bf;
        bp[0] = *(const v8h*)(brow + k0 + ks);
        bp[1] = *(const v8h*)(brow + k0 + ks + 16);
        v8h* xp = (v8h*)&xf;
        xp[0] = *(const v8h*)(&Xh[fr * 64 + ks + akb]);
        xp[1] = *(const v8h*)(&Xh[fr * 64 + ks + 16 + akb]);
        v8h* sp = (v8h*)&sf;
        sp[0] = *(const v8h*)(&Sh[fr * 64 + ks + akb]);
        sp[1] = *(const v8h*)(&Sh[fr * 64 + ks + 16 + akb]);
      }
      acc1 = __builtin_amdgcn_wmma_f32_16x16x32_f16(false, sf, false, bf,
                                                    (short)0, acc1, false, false);
      acc2 = __builtin_amdgcn_wmma_f32_16x16x32_f16(false, xf, false, bf,
                                                    (short)0, acc2, false, false);
    }
    __syncthreads();
  }

  // ---- store C tiles (C/D layout: vgpr i -> row i + (lane>>4)*8, col lane&15)
  const int crow = (lane >> 4) * 8;
  const int ccol = lane & 15;
  const size_t obase = (size_t)(mblk * 16 + crow) * MTGT + n0 + ccol;
  #pragma unroll
  for (int i = 0; i < 8; ++i) {
    C1[obase + (size_t)i * MTGT] = acc1[i];
    C2[obase + (size_t)i * MTGT] = acc2[i];
  }

  // ---- per-pred-row sums (only column strip 0 contributes) ----
  if (cstrip == 0) {
    redS[tid] = ps;
    redL[tid] = pl;
    __syncthreads();
    if ((tid & 15) == 0) {
      float a = 0.0f, b = 0.0f;
      #pragma unroll
      for (int i = 0; i < 16; ++i) { a += redS[tid + i]; b += redL[tid + i]; }
      ssum[mblk * 16 + rowA] = a;
      lsum[mblk * 16 + rowA] = b;
    }
  }
}

// ---------------------------------------------------------------------------
// Kernel 3: combine all cost terms into C[1200][256].
// ---------------------------------------------------------------------------
__global__ __launch_bounds__(256) void k_combine(
    const float* __restrict__ logits, const float* __restrict__ pboxes,
    const int* __restrict__ tids, const float* __restrict__ tboxes,
    const float* __restrict__ C1, const float* __restrict__ C2,
    const float* __restrict__ ssum, const float* __restrict__ lsum,
    const float* __restrict__ tsum, float* __restrict__ out) {
  const int n = blockIdx.x;     // pred index 0..1199
  const int m = threadIdx.x;    // target index 0..255

  // class cost: -sigmoid(logit[tgt_id])
  const int id = tids[m];
  const float lg = logits[(size_t)n * NCLS + id];
  const float cost_class = -(1.0f / (1.0f + __expf(-lg)));

  // bbox L1 + GIoU
  const float pcx = pboxes[n * 4 + 0], pcy = pboxes[n * 4 + 1];
  const float pw  = pboxes[n * 4 + 2], phh = pboxes[n * 4 + 3];
  const float tcx = tboxes[m * 4 + 0], tcy = tboxes[m * 4 + 1];
  const float tw  = tboxes[m * 4 + 2], thh = tboxes[m * 4 + 3];
  const float cost_bbox = fabsf(pcx - tcx) + fabsf(pcy - tcy) +
                          fabsf(pw - tw) + fabsf(phh - thh);

  const float px1 = pcx - 0.5f * pw,  py1 = pcy - 0.5f * phh;
  const float px2 = pcx + 0.5f * pw,  py2 = pcy + 0.5f * phh;
  const float tx1 = tcx - 0.5f * tw,  ty1 = tcy - 0.5f * thh;
  const float tx2 = tcx + 0.5f * tw,  ty2 = tcy + 0.5f * thh;
  const float a1 = (px2 - px1) * (py2 - py1);
  const float a2 = (tx2 - tx1) * (ty2 - ty1);
  const float iw = fmaxf(fminf(px2, tx2) - fmaxf(px1, tx1), 0.0f);
  const float ih = fmaxf(fminf(py2, ty2) - fmaxf(py1, ty1), 0.0f);
  const float inter = iw * ih;
  const float uni = a1 + a2 - inter;
  const float iou = inter / uni;
  const float ew = fmaxf(fmaxf(px2, tx2) - fminf(px1, tx1), 0.0f);
  const float eh = fmaxf(fmaxf(py2, ty2) - fminf(py1, ty1), 0.0f);
  const float ae = ew * eh;
  const float cost_giou = -(iou - (ae - uni) / ae);

  // mask costs from the two GEMMs
  const size_t idx = (size_t)n * MTGT + m;
  const float num = 2.0f * C1[idx];
  const float den = ssum[n] + tsum[m];
  const float cost_dice = -(num + 1.0f) / (fmaxf(den, 1e-6f) + 1.0f);
  const float cost_mask = -(C2[idx] + lsum[n]) * (1.0f / (float)KP);

  out[idx] = 5.0f * cost_bbox + 2.0f * cost_giou + 2.0f * cost_class +
             2.0f * cost_mask + 2.0f * cost_dice;
}

// ---------------------------------------------------------------------------
extern "C" void kernel_launch(void* const* d_in, const int* in_sizes, int n_in,
                              void* d_out, int out_size, void* d_ws, size_t ws_size,
                              hipStream_t stream) {
  const float* pred_logits = (const float*)d_in[0];   // [4,300,80]
  const float* pred_boxes  = (const float*)d_in[1];   // [4,300,4]
  const float* pred_masks  = (const float*)d_in[2];   // [4,300,128,128]
  const int*   tgt_ids     = (const int*)d_in[3];     // [256]
  const float* tgt_boxes   = (const float*)d_in[4];   // [256,4]
  const float* tgt_masks   = (const float*)d_in[5];   // [256,256,256]
  float* out = (float*)d_out;                         // [4,300,256]

  char* ws = (char*)d_ws;
  _Float16* T = (_Float16*)(ws);                      // 256*16384*2  = 8,388,608 B
  float* C1   = (float*)(ws + 8388608);               // 1200*256*4   = 1,228,800 B
  float* C2   = (float*)(ws + 8388608 + 1228800);     // 1,228,800 B
  float* tsum = (float*)(ws + 8388608 + 2 * 1228800); // 256*4
  float* ssum = tsum + MTGT;                          // 1200*4
  float* lsum = ssum + NPRED;                         // 1200*4

  hipLaunchKernelGGL(k_prep_tgt, dim3(MTGT), dim3(256), 0, stream,
                     tgt_masks, T, tsum);
  hipLaunchKernelGGL(k_gemm, dim3(75, 2), dim3(256), 0, stream,
                     pred_masks, T, C1, C2, ssum, lsum);
  hipLaunchKernelGGL(k_combine, dim3(NPRED), dim3(256), 0, stream,
                     pred_logits, pred_boxes, tgt_ids, tgt_boxes,
                     C1, C2, ssum, lsum, tsum, out);
}
